// STA_LSTM_88725434401479
// MI455X (gfx1250) — compile-verified, matrix operand order checked by reading, not run on hardware
//
#include <hip/hip_runtime.h>
#include <hip/hip_bf16.h>

typedef __bf16 bf16;
typedef __attribute__((ext_vector_type(16))) __bf16 v16bf;
typedef __attribute__((ext_vector_type(8)))  float  v8f;
typedef __attribute__((ext_vector_type(4)))  float  v4f;
typedef __attribute__((ext_vector_type(4)))  unsigned int u32x4;
typedef __attribute__((ext_vector_type(4)))  int  i32x4;
typedef __attribute__((ext_vector_type(8)))  int  i32x8;

constexpr int Bn  = 2048;   // batch
constexpr int Sn  = 128;    // sequence
constexpr int Dn  = 64;     // input feature
constexpr int H1n = 128;    // sa hidden
constexpr int H2n = 128;    // ta hidden
constexpr int G1n = 4 * H1n;  // 512
constexpr int G2n = 4 * H2n;  // 512
constexpr int MT  = 16;     // batch tile per workgroup (== WMMA M)

#if defined(__has_builtin)
#if __has_builtin(__builtin_amdgcn_tensor_load_to_lds) && __has_builtin(__builtin_amdgcn_s_wait_tensorcnt)
#define STA_HAVE_TDM 1
#endif
#endif

__device__ __forceinline__ float sigmoidf_(float x) { return 1.0f / (1.0f + expf(-x)); }

__device__ __forceinline__ v8f wmma_bf16(v16bf a, v16bf b, v8f c) {
  return __builtin_amdgcn_wmma_f32_16x16x32_bf16(false, a, false, b, (short)0, c, false, false);
}

// A fragment (16x32 bf16) from LDS, row-major MxK with leading dim lda.
// ISA 7.12.2: lanes 0-15 -> K=0..7 (v0-3), K=16..23 (v4-7); lanes 16-31 -> K=8..15, K=24..31.
__device__ __forceinline__ v16bf load_a(const bf16* __restrict__ A, int lda, int k0, int lane) {
  const int m = lane & 15, hi = lane >> 4;
  const bf16* p = A + m * lda + k0 + 8 * hi;
  v16bf a;
#pragma unroll
  for (int e = 0; e < 8; ++e) a[e] = p[e];
#pragma unroll
  for (int e = 0; e < 8; ++e) a[8 + e] = p[16 + e];
  return a;
}

// B fragment from pre-packed bf16 weights: tile (kt, nt) of 32x16, lane-major,
// each lane's 16 bf16 contiguous (32 bytes) -> one 32B contiguous load (L2-resident).
__device__ __forceinline__ v16bf load_bp(const bf16* __restrict__ P, int KT, int kt, int nt, int lane) {
  return *(const v16bf*)(P + (((size_t)(nt * KT + kt) * 32 + lane) << 4));
}

#ifdef STA_HAVE_TDM
// TDM: DMA a 2D f32 tile (tile_y rows x tile_x elems, row stride in elems) into LDS.
// Descriptor per CDNA5 ISA sec 8 (D# group0/group1); groups 2/3 unused (2D tensor).
__device__ __forceinline__ void tdm_load_2d_f32(unsigned lds_addr, const float* gsrc,
                                                int tile_x, int tile_y, long long stride_elems) {
  const unsigned long long ga = (unsigned long long)(size_t)gsrc;
  u32x4 g0;
  g0[0] = 1u;                                               // count=1, user descriptor
  g0[1] = lds_addr;                                         // LDS byte address
  g0[2] = (unsigned)(ga & 0xFFFFFFFFu);                     // global addr lo
  g0[3] = (unsigned)((ga >> 32) & 0x01FFFFFFu) | (2u << 30);// global addr hi | type=2
  i32x8 g1;
  g1[0] = (int)(2u << 16);                                  // workgroup_mask=0, data_size=2 (4B)
  g1[1] = (int)(((unsigned)tile_x & 0xFFFFu) << 16);        // tensor_dim0[15:0]
  g1[2] = (int)(((unsigned)tile_y & 0xFFFFu) << 16);        // tensor_dim0 hi=0 | tensor_dim1 lo
  g1[3] = (int)(((unsigned)tile_x & 0xFFFFu) << 16);        // tensor_dim1 hi=0 | tile_dim0
  g1[4] = (int)((unsigned)tile_y & 0xFFFFu);                // tile_dim1 | tile_dim2=0
  g1[5] = (int)(unsigned)(stride_elems & 0xFFFFFFFFll);     // tensor_dim0_stride lo32
  g1[6] = (int)(unsigned)((stride_elems >> 32) & 0xFFFFll); // stride hi16 | dim1_stride lo=0
  g1[7] = 0;
  i32x4 z4 = {};
#if __clang_major__ >= 23
  i32x8 z8 = {};
  __builtin_amdgcn_tensor_load_to_lds(g0, g1, z4, z4, z8, 0);
#else
  __builtin_amdgcn_tensor_load_to_lds(g0, g1, z4, z4, 0);
#endif
}
#endif

// ---------------- one-time weight pack: fp32 (KxN) -> bf16 WMMA-B fragment layout ----------------
__global__ __launch_bounds__(256) void sta_pack_kernel(const float* __restrict__ W, int K, int N,
                                                       bf16* __restrict__ P) {
  const int idx = blockIdx.x * 256 + threadIdx.x;   // one lane-slot (16 bf16) per thread
  const int KT = K >> 5, NT = N >> 4;
  if (idx >= KT * NT * 32) return;
  const int lane = idx & 31, tile = idx >> 5;
  const int kt = tile % KT, nt = tile / KT;
  const int n = nt * 16 + (lane & 15), hi = lane >> 4;
  bf16* dst = P + ((size_t)idx << 4);
#pragma unroll
  for (int e = 0; e < 8; ++e) dst[e]     = (bf16)W[(size_t)(kt * 32 + 8 * hi + e) * N + n];
#pragma unroll
  for (int e = 0; e < 8; ++e) dst[8 + e] = (bf16)W[(size_t)(kt * 32 + 16 + 8 * hi + e) * N + n];
}

// ---------------- Spatial attention LSTM scan ----------------
__global__ __launch_bounds__(256) void sta_sa_kernel(
    const float* __restrict__ X,
    const bf16* __restrict__ pW,  const bf16* __restrict__ pU,   // packed (D,4H1),(H1,4H1)
    const float* __restrict__ bG,
    const bf16* __restrict__ pWa, const bf16* __restrict__ pUa,  // packed (D,D),(2H1,D)
    const float* __restrict__ bA, const bf16* __restrict__ pVa,  // (D), packed (D,D)
    float* __restrict__ alphas_out, float* __restrict__ Hws) {
  __shared__ float xt_f[MT * Dn];
  __shared__ bf16  xt_bf[MT * Dn];
  __shared__ bf16  hc_bf[MT * (2 * H1n)];   // [h | c] per row, stride 256
  __shared__ bf16  t1_bf[MT * Dn];
  __shared__ float a_f[MT * Dn];
  __shared__ bf16  xw_bf[MT * Dn];
  __shared__ float c_f[MT * H1n];

  const int tid = threadIdx.x, lane = tid & 31, wv = tid >> 5;
  const int b0 = blockIdx.x * MT;

  for (int i = tid; i < MT * 2 * H1n; i += 256) hc_bf[i] = (bf16)0.0f;
  for (int i = tid; i < MT * H1n; i += 256) c_f[i] = 0.0f;
  __syncthreads();

  for (int t = 0; t < Sn; ++t) {
    // ---- stage x_t tile (16 x 64 f32, row stride S*D) ----
#ifdef STA_HAVE_TDM
    if (wv == 0) {
      tdm_load_2d_f32((unsigned)(size_t)xt_f, X + (size_t)b0 * (Sn * Dn) + (size_t)t * Dn,
                      Dn, MT, (long long)(Sn * Dn));
      __builtin_amdgcn_s_wait_tensorcnt(0);
    }
    __syncthreads();
    for (int i = tid; i < MT * Dn; i += 256) xt_bf[i] = (bf16)xt_f[i];
#else
    for (int i = tid; i < MT * Dn; i += 256) {
      const int r = i >> 6, d = i & (Dn - 1);
      const float v = X[(size_t)(b0 + r) * (Sn * Dn) + (size_t)t * Dn + d];
      xt_f[i] = v; xt_bf[i] = (bf16)v;
    }
#endif
    __syncthreads();

    // ---- stage 1: t1 = tanh(x_t @ Wa + [h,c] @ Ua + bA)  (16 x 64, 4 n-tiles) ----
    if (wv < 4) {
      const int col = wv * 16 + (lane & 15), hi = lane >> 4;
      v8f acc;
      const float bias = bA[col];
#pragma unroll
      for (int e = 0; e < 8; ++e) acc[e] = bias;
#pragma unroll
      for (int kt = 0; kt < 2; ++kt)
        acc = wmma_bf16(load_a(xt_bf, Dn, kt * 32, lane), load_bp(pWa, 2, kt, wv, lane), acc);
#pragma unroll
      for (int kt = 0; kt < 8; ++kt)
        acc = wmma_bf16(load_a(hc_bf, 2 * H1n, kt * 32, lane), load_bp(pUa, 8, kt, wv, lane), acc);
#pragma unroll
      for (int e = 0; e < 8; ++e) t1_bf[(e + 8 * hi) * Dn + col] = (bf16)tanhf(acc[e]);
    }
    __syncthreads();

    // ---- stage 2: a = t1 @ Va ----
    if (wv < 4) {
      const int col = wv * 16 + (lane & 15), hi = lane >> 4;
      v8f acc = {};
#pragma unroll
      for (int kt = 0; kt < 2; ++kt)
        acc = wmma_bf16(load_a(t1_bf, Dn, kt * 32, lane), load_bp(pVa, 2, kt, wv, lane), acc);
#pragma unroll
      for (int e = 0; e < 8; ++e) a_f[(e + 8 * hi) * Dn + col] = acc[e];
    }
    __syncthreads();

    // ---- stage 3: softmax over D=64, emit alpha, build xw = alpha * x_t ----
    {
      const int r = tid >> 4, i2 = tid & 15;  // 16 lanes per row, 4 cols each
      float v[4];
#pragma unroll
      for (int j = 0; j < 4; ++j) v[j] = a_f[r * Dn + i2 + 16 * j];
      float mx = fmaxf(fmaxf(v[0], v[1]), fmaxf(v[2], v[3]));
      for (int off = 8; off > 0; off >>= 1) mx = fmaxf(mx, __shfl_xor(mx, off, 16));
      float sum = 0.0f;
#pragma unroll
      for (int j = 0; j < 4; ++j) { v[j] = expf(v[j] - mx); sum += v[j]; }
      for (int off = 8; off > 0; off >>= 1) sum += __shfl_xor(sum, off, 16);
      const float inv = 1.0f / sum;
#pragma unroll
      for (int j = 0; j < 4; ++j) {
        const int d = i2 + 16 * j;
        const float al = v[j] * inv;
        alphas_out[((size_t)t * Bn + (b0 + r)) * Dn + d] = al;
        xw_bf[r * Dn + d] = (bf16)(al * xt_f[r * Dn + d]);
      }
    }
    __syncthreads();

    // ---- stage 4: gates = xw @ W + h @ U + bG ; LSTM update (wave w owns h-cols 16w..16w+15) ----
    {
      const int colL = lane & 15, hi = lane >> 4;
      const int nI = wv, nF = wv + 8, nG = wv + 16, nO = wv + 24;
      v8f gi, gf, gg, go;
#pragma unroll
      for (int e = 0; e < 8; ++e) {
        gi[e] = bG[nI * 16 + colL]; gf[e] = bG[nF * 16 + colL];
        gg[e] = bG[nG * 16 + colL]; go[e] = bG[nO * 16 + colL];
      }
#pragma unroll
      for (int kt = 0; kt < 2; ++kt) {
        const v16bf a = load_a(xw_bf, Dn, kt * 32, lane);
        gi = wmma_bf16(a, load_bp(pW, 2, kt, nI, lane), gi);
        gf = wmma_bf16(a, load_bp(pW, 2, kt, nF, lane), gf);
        gg = wmma_bf16(a, load_bp(pW, 2, kt, nG, lane), gg);
        go = wmma_bf16(a, load_bp(pW, 2, kt, nO, lane), go);
      }
#pragma unroll
      for (int kt = 0; kt < 4; ++kt) {
        const v16bf a = load_a(hc_bf, 2 * H1n, kt * 32, lane);  // h part (cols 0..127)
        gi = wmma_bf16(a, load_bp(pU, 4, kt, nI, lane), gi);
        gf = wmma_bf16(a, load_bp(pU, 4, kt, nF, lane), gf);
        gg = wmma_bf16(a, load_bp(pU, 4, kt, nG, lane), gg);
        go = wmma_bf16(a, load_bp(pU, 4, kt, nO, lane), go);
      }
      __syncthreads();  // all waves done READING old h before anyone writes new h
      const int hcol = wv * 16 + colL;
#pragma unroll
      for (int e = 0; e < 8; ++e) {
        const int m = e + 8 * hi;
        const float iV = sigmoidf_(gi[e]), fV = sigmoidf_(gf[e]);
        const float gV = tanhf(gg[e]),    oV = sigmoidf_(go[e]);
        const float cN = fV * c_f[m * H1n + hcol] + iV * gV;
        const float hN = oV * tanhf(cN);
        c_f[m * H1n + hcol] = cN;
        hc_bf[m * (2 * H1n) + hcol] = (bf16)hN;
        hc_bf[m * (2 * H1n) + H1n + hcol] = (bf16)cN;
        Hws[(size_t)(b0 + m) * (Sn * H1n) + (size_t)t * H1n + hcol] = hN;
      }
    }
    __syncthreads();
  }
}

// ---------------- Temporal attention LSTM scan ----------------
__global__ __launch_bounds__(256) void sta_ta_kernel(
    const float* __restrict__ Hws, const float* __restrict__ X,
    const bf16* __restrict__ pWa,  const float* __restrict__ bA,  // packed (H1,S),(S)
    const bf16* __restrict__ pVa,                                 // packed (S,S)
    const bf16* __restrict__ pW,   const bf16* __restrict__ pU,   // packed (H1,4H2),(H2,4H2)
    const float* __restrict__ bG,  const float* __restrict__ Wy,  // (4H2),(1,4H2)
    const float* __restrict__ fcw, const float* __restrict__ fcb, // (1,H2),(1)
    float* __restrict__ betas_out, float* __restrict__ y_out) {
  __shared__ float ht_f[MT * H1n];
  __shared__ bf16  ht_bf[MT * H1n];
  __shared__ bf16  t1_bf[MT * Sn];
  __shared__ float a_f[MT * Sn];
  __shared__ float beta_f[MT * Sn];
  __shared__ bf16  hatt_bf[MT * H1n];
  __shared__ bf16  lh_bf[MT * H2n];
  __shared__ float lc_f[MT * H2n];
  __shared__ float y_f[MT];

  const int tid = threadIdx.x, lane = tid & 31, wv = tid >> 5;
  const int b0 = blockIdx.x * MT;

  for (int i = tid; i < MT * H2n; i += 256) { lh_bf[i] = (bf16)0.0f; lc_f[i] = 0.0f; }
  if (tid < MT) y_f[tid] = X[(size_t)(b0 + tid) * (Sn * Dn) + (Dn - 1)];  // y0 = X[:,0,D-1]
  __syncthreads();

  for (int t = 0; t < Sn; ++t) {
    // ---- stage h_t (16 x 128 f32, row stride S*H1) ----
#ifdef STA_HAVE_TDM
    if (wv == 0) {
      tdm_load_2d_f32((unsigned)(size_t)ht_f, Hws + (size_t)b0 * (Sn * H1n) + (size_t)t * H1n,
                      H1n, MT, (long long)(Sn * H1n));
      __builtin_amdgcn_s_wait_tensorcnt(0);
    }
    __syncthreads();
    for (int i = tid; i < MT * H1n; i += 256) ht_bf[i] = (bf16)ht_f[i];
#else
    for (int i = tid; i < MT * H1n; i += 256) {
      const int r = i >> 7, d = i & (H1n - 1);
      ht_bf[i] = (bf16)Hws[(size_t)(b0 + r) * (Sn * H1n) + (size_t)t * H1n + d];
    }
#endif
    __syncthreads();

    // ---- t1 = tanh(h_t @ Wa + bA) ----
    {
      const int col = wv * 16 + (lane & 15), hi = lane >> 4;
      v8f acc;
      const float bias = bA[col];
#pragma unroll
      for (int e = 0; e < 8; ++e) acc[e] = bias;
#pragma unroll
      for (int kt = 0; kt < 4; ++kt)
        acc = wmma_bf16(load_a(ht_bf, H1n, kt * 32, lane), load_bp(pWa, 4, kt, wv, lane), acc);
#pragma unroll
      for (int e = 0; e < 8; ++e) t1_bf[(e + 8 * hi) * Sn + col] = (bf16)tanhf(acc[e]);
    }
    __syncthreads();

    // ---- a = t1 @ Va ----
    {
      const int col = wv * 16 + (lane & 15), hi = lane >> 4;
      v8f acc = {};
#pragma unroll
      for (int kt = 0; kt < 4; ++kt)
        acc = wmma_bf16(load_a(t1_bf, Sn, kt * 32, lane), load_bp(pVa, 4, kt, wv, lane), acc);
#pragma unroll
      for (int e = 0; e < 8; ++e) a_f[(e + 8 * hi) * Sn + col] = acc[e];
    }
    __syncthreads();

    // ---- softmax over S=128, emit beta ----
    {
      const int r = tid >> 4, i2 = tid & 15;  // 8 cols each
      float v[8]; float mx = -3.0e38f;
#pragma unroll
      for (int j = 0; j < 8; ++j) { v[j] = a_f[r * Sn + i2 + 16 * j]; mx = fmaxf(mx, v[j]); }
      for (int off = 8; off > 0; off >>= 1) mx = fmaxf(mx, __shfl_xor(mx, off, 16));
      float sum = 0.0f;
#pragma unroll
      for (int j = 0; j < 8; ++j) { v[j] = expf(v[j] - mx); sum += v[j]; }
      for (int off = 8; off > 0; off >>= 1) sum += __shfl_xor(sum, off, 16);
      const float inv = 1.0f / sum;
#pragma unroll
      for (int j = 0; j < 8; ++j) {
        const int s = i2 + 16 * j;
        const float bv = v[j] * inv;
        beta_f[r * Sn + s] = bv;
        betas_out[((size_t)t * Bn + (b0 + r)) * Sn + s] = bv;
      }
    }
    __syncthreads();

    // ---- h_att[r,d] = sum_s beta[r,s] * H[b_r, s, d]  (per-sample, VALU, float4, L2-served) ----
    {
      const int c4 = (tid & 31) * 4;   // column group of 4
      const int r0 = tid >> 5;         // rows r0 and r0+8
      v4f acc0 = {}, acc1 = {};
      const float* H0 = Hws + (size_t)(b0 + r0) * (Sn * H1n) + c4;
      const float* H1 = Hws + (size_t)(b0 + r0 + 8) * (Sn * H1n) + c4;
      for (int s = 0; s < Sn; ++s) {
        const v4f hv0 = *(const v4f*)(H0 + (size_t)s * H1n);
        const v4f hv1 = *(const v4f*)(H1 + (size_t)s * H1n);
        acc0 += hv0 * beta_f[r0 * Sn + s];
        acc1 += hv1 * beta_f[(r0 + 8) * Sn + s];
      }
#pragma unroll
      for (int j = 0; j < 4; ++j) {
        hatt_bf[r0 * H1n + c4 + j]       = (bf16)acc0[j];
        hatt_bf[(r0 + 8) * H1n + c4 + j] = (bf16)acc1[j];
      }
    }
    __syncthreads();

    // ---- gates = h_att @ W + lh @ U + y_prev * Wy + bG ; LSTM update ----
    {
      const int colL = lane & 15, hi = lane >> 4;
      const int nI = wv, nF = wv + 8, nG = wv + 16, nO = wv + 24;
      v8f gi, gf, gg, go;
#pragma unroll
      for (int e = 0; e < 8; ++e) {
        const float yp = y_f[e + 8 * hi];
        gi[e] = bG[nI * 16 + colL] + yp * Wy[nI * 16 + colL];
        gf[e] = bG[nF * 16 + colL] + yp * Wy[nF * 16 + colL];
        gg[e] = bG[nG * 16 + colL] + yp * Wy[nG * 16 + colL];
        go[e] = bG[nO * 16 + colL] + yp * Wy[nO * 16 + colL];
      }
#pragma unroll
      for (int kt = 0; kt < 4; ++kt) {
        const v16bf a = load_a(hatt_bf, H1n, kt * 32, lane);
        gi = wmma_bf16(a, load_bp(pW, 4, kt, nI, lane), gi);
        gf = wmma_bf16(a, load_bp(pW, 4, kt, nF, lane), gf);
        gg = wmma_bf16(a, load_bp(pW, 4, kt, nG, lane), gg);
        go = wmma_bf16(a, load_bp(pW, 4, kt, nO, lane), go);
      }
#pragma unroll
      for (int kt = 0; kt < 4; ++kt) {
        const v16bf a = load_a(lh_bf, H2n, kt * 32, lane);
        gi = wmma_bf16(a, load_bp(pU, 4, kt, nI, lane), gi);
        gf = wmma_bf16(a, load_bp(pU, 4, kt, nF, lane), gf);
        gg = wmma_bf16(a, load_bp(pU, 4, kt, nG, lane), gg);
        go = wmma_bf16(a, load_bp(pU, 4, kt, nO, lane), go);
      }
      __syncthreads();  // all waves done READING old lh before overwrite
      const int hcol = wv * 16 + colL;
#pragma unroll
      for (int e = 0; e < 8; ++e) {
        const int m = e + 8 * hi;
        const float iV = sigmoidf_(gi[e]), fV = sigmoidf_(gf[e]);
        const float gV = tanhf(gg[e]),    oV = sigmoidf_(go[e]);
        const float cN = fV * lc_f[m * H2n + hcol] + iV * gV;
        const float hN = oV * tanhf(cN);
        lc_f[m * H2n + hcol] = cN;
        lh_bf[m * H2n + hcol] = (bf16)hN;
      }
    }
    __syncthreads();

    // ---- y = lh @ fc_w^T + fc_b ----
    if (tid < 128) {
      const int r = tid >> 3, j = tid & 7;  // 8 lanes per row
      float sum = 0.0f;
#pragma unroll
      for (int k = 0; k < 16; ++k) {
        const int d2 = j + 8 * k;
        sum = fmaf((float)lh_bf[r * H2n + d2], fcw[d2], sum);
      }
      for (int off = 4; off > 0; off >>= 1) sum += __shfl_xor(sum, off, 8);
      if (j == 0) {
        const float yv = sum + fcb[0];
        y_f[r] = yv;
        if (t == Sn - 1) y_out[b0 + r] = yv;
      }
    }
    __syncthreads();
  }
}

extern "C" void kernel_launch(void* const* d_in, const int* in_sizes, int n_in,
                              void* d_out, int out_size, void* d_ws, size_t ws_size,
                              hipStream_t stream) {
  const float* X    = (const float*)d_in[0];
  const float* saW  = (const float*)d_in[1];
  const float* saU  = (const float*)d_in[2];
  const float* sab  = (const float*)d_in[3];
  const float* saWa = (const float*)d_in[4];
  const float* saUa = (const float*)d_in[5];
  const float* saba = (const float*)d_in[6];
  const float* saVa = (const float*)d_in[7];
  const float* taWa = (const float*)d_in[8];
  // d_in[9] = ta_Ua : present in inputs but unused by the reference forward pass
  const float* taba = (const float*)d_in[10];
  const float* taVa = (const float*)d_in[11];
  const float* taW  = (const float*)d_in[12];
  const float* taU  = (const float*)d_in[13];
  const float* tab  = (const float*)d_in[14];
  const float* taWy = (const float*)d_in[15];
  const float* fcw  = (const float*)d_in[16];
  const float* fcb  = (const float*)d_in[17];

  float* out    = (float*)d_out;
  float* y_out  = out;                                   // B * OUT
  float* alphas = out + Bn;                              // S*B*D
  float* betas  = alphas + (size_t)Sn * Bn * Dn;         // S*B*S

  // workspace: [ H history fp32 | packed bf16 weights ]
  float* Hws = (float*)d_ws;
  bf16* pk = (bf16*)((char*)d_ws + (size_t)Bn * Sn * H1n * sizeof(float));
  bf16* p_saWa = pk; pk += 64 * 64;
  bf16* p_saUa = pk; pk += 256 * 64;
  bf16* p_saVa = pk; pk += 64 * 64;
  bf16* p_saW  = pk; pk += 64 * 512;
  bf16* p_saU  = pk; pk += 128 * 512;
  bf16* p_taWa = pk; pk += 128 * 128;
  bf16* p_taVa = pk; pk += 128 * 128;
  bf16* p_taW  = pk; pk += 128 * 512;
  bf16* p_taU  = pk; pk += 128 * 512;

  auto pack = [&](const float* src, int K, int N, bf16* dst) {
    const int total = (K >> 5) * (N >> 4) * 32;
    sta_pack_kernel<<<(total + 255) / 256, 256, 0, stream>>>(src, K, N, dst);
  };
  pack(saWa, 64, 64, p_saWa);
  pack(saUa, 256, 64, p_saUa);
  pack(saVa, 64, 64, p_saVa);
  pack(saW, 64, 512, p_saW);
  pack(saU, 128, 512, p_saU);
  pack(taWa, 128, 128, p_taWa);
  pack(taVa, 128, 128, p_taVa);
  pack(taW, 128, 512, p_taW);
  pack(taU, 128, 512, p_taU);

  dim3 grid(Bn / MT), blk(256);
  sta_sa_kernel<<<grid, blk, 0, stream>>>(X, p_saW, p_saU, sab, p_saWa, p_saUa, saba, p_saVa,
                                          alphas, Hws);
  sta_ta_kernel<<<grid, blk, 0, stream>>>(Hws, X, p_taWa, taba, p_taVa, p_taW, p_taU, tab, taWy,
                                          fcw, fcb, betas, y_out);
}
